// Base_70411693850975
// MI455X (gfx1250) — matrix operand order repaired and checked by reading, not compiled, not run on hardware
//
#include <hip/hip_runtime.h>
#include <hip/hip_bf16.h>
#include <math.h>

typedef __attribute__((ext_vector_type(16))) _Float16 v16h;
typedef __attribute__((ext_vector_type(2)))  _Float16 h2;
typedef __attribute__((ext_vector_type(8)))  float    v8f;

#define N_RAYS      8192
#define N_SAMPLES   259              // int(sqrt(27)/STEP)+1
#define STEP        (6.0f/299.0f)    // mean(units)*STEP_RATIO
#define NEARP       2.0f
#define FARP        6.0f
#define DIST_SCALE  25.0f

#define RAYS_PER_WAVE   16
#define WAVES_PER_BLOCK 4
#define THREADS_PER_BLOCK (WAVES_PER_BLOCK * 32)
#define RAYS_PER_BLOCK  (RAYS_PER_WAVE * WAVES_PER_BLOCK)   // 64
#define NBLOCKS         (N_RAYS / RAYS_PER_BLOCK)           // 128

static __device__ __forceinline__ void lds_fence() {
  // wave-local LDS RAW ordering (CDNA5 split counters) + compiler mem barrier
  asm volatile("s_wait_dscnt 0x0" ::: "memory");
}

// Load a 16x32 f16 A-fragment from an LDS tile laid out [16 rows][ldk K].
// Contiguous-k A layout used here (B frags match): lanes 0-15 hold row M=lane, VGPR v pairs:
//   K = 2v + 16*hi  (+ kstep*32), hi = lane>>4
static __device__ __forceinline__ v16h load_a_frag(const _Float16* tile, int ldk,
                                                   int lane, int kstep) {
  const int row = lane & 15;
  const int hi  = lane >> 4;
  const _Float16* p = tile + row * ldk + kstep * 32 + 16 * hi;
  v16h a;
#pragma unroll
  for (int v = 0; v < 8; ++v) {
    const int kb = 2 * v;
    h2 pr = *(const h2*)(p + kb);
    a[2 * v]     = pr.x;
    a[2 * v + 1] = pr.y;
  }
  return a;
}

__global__ __launch_bounds__(THREADS_PER_BLOCK)
void nerf_wmma_kernel(const float* __restrict__ rays,   // [8192][6]
                      const float* __restrict__ W1,     // [36][128]
                      const float* __restrict__ b1,     // [128]
                      const float* __restrict__ w_d,    // [128]
                      const float* __restrict__ W_rgb,  // [131][3]
                      float* __restrict__ out)          // rgb[8192*3] | depth[8192] | acc[8192]
{
  // Shared weight tiles (block-wide), f16, transposed for contiguous B frags.
  __shared__ __align__(32) _Float16 W1T[128][64];   // [N][K], K padded 36->64
  __shared__ __align__(32) _Float16 W2T[16][128];   // [N][K]: n0=w_d, n1..3=W_rgb cols, rest 0
  // Per-wave staging tiles.
  __shared__ __align__(32) _Float16 peT[WAVES_PER_BLOCK][16][64];    // A tile, layer 1
  __shared__ __align__(32) _Float16 featT[WAVES_PER_BLOCK][16][128]; // A tile, head
  __shared__ __align__(16) float    outT[WAVES_PER_BLOCK][16][4];    // sigma,r,g,b raw

  const int tid = threadIdx.x;

  // ---- Cooperative fill of weight tiles (f32 -> f16, transpose, zero-pad) ----
  for (int i = tid; i < 128 * 64; i += THREADS_PER_BLOCK) {
    const int n = i >> 6, k = i & 63;
    const float v = (k < 36) ? W1[k * 128 + n] : 0.0f;
    W1T[n][k] = (_Float16)v;
  }
  for (int i = tid; i < 16 * 128; i += THREADS_PER_BLOCK) {
    const int n = i >> 7, k = i & 127;
    float v = 0.0f;
    if (n == 0)      v = w_d[k];
    else if (n < 4)  v = W_rgb[k * 3 + (n - 1)];
    W2T[n][k] = (_Float16)v;
  }
  __syncthreads();

  const int wave = tid >> 5;
  const int lane = tid & 31;
  const int row  = lane & 15;     // M row / N col index within a 16-tile
  const int hi   = lane >> 4;
  const int ray  = blockIdx.x * RAYS_PER_BLOCK + wave * RAYS_PER_WAVE + row;

  // Zero the PE tile once; only k<36 gets overwritten each sample.
  for (int i = lane; i < 16 * 64; i += 32)
    peT[wave][i >> 6][i & 63] = (_Float16)0.0f;

  // ---- Per-ray setup (lanes 16-31 mirror lanes 0-15's ray) ----
  const float ox = rays[ray * 6 + 0], oy = rays[ray * 6 + 1], oz = rays[ray * 6 + 2];
  const float dx = rays[ray * 6 + 3], dy = rays[ray * 6 + 4], dz = rays[ray * 6 + 5];
  const float vx = (dx == 0.0f) ? 1e-6f : dx;
  const float vy = (dy == 0.0f) ? 1e-6f : dy;
  const float vz = (dz == 0.0f) ? 1e-6f : dz;
  const float tx = fminf(( 1.5f - ox) / vx, (-1.5f - ox) / vx);
  const float ty = fminf(( 1.5f - oy) / vy, (-1.5f - oy) / vy);
  const float tz = fminf(( 1.5f - oz) / vz, (-1.5f - oz) / vz);
  float t_min = fmaxf(fmaxf(tx, ty), tz);
  t_min = fminf(fmaxf(t_min, NEARP), FARP);

  // Layer-1 bias per N column (this lane's column), for each of 8 N-tiles.
  float biasv[8];
#pragma unroll
  for (int nt = 0; nt < 8; ++nt) biasv[nt] = b1[nt * 16 + row];

  // Head B fragments (loop-invariant): col = row, lanes hi=1 take K+16.
  v16h b2[4];
#pragma unroll
  for (int ks = 0; ks < 4; ++ks)
    b2[ks] = *(const v16h*)&W2T[row][ks * 32 + hi * 16];

  // Viewdir contribution to rgb raw (per-ray constant).
  const float vt0 = dx * W_rgb[128 * 3 + 0] + dy * W_rgb[129 * 3 + 0] + dz * W_rgb[130 * 3 + 0];
  const float vt1 = dx * W_rgb[128 * 3 + 1] + dy * W_rgb[129 * 3 + 1] + dz * W_rgb[130 * 3 + 1];
  const float vt2 = dx * W_rgb[128 * 3 + 2] + dy * W_rgb[129 * 3 + 2] + dz * W_rgb[130 * 3 + 2];

  // Compositing state (lanes 0-15 own a ray each).
  float T = 1.0f, accw = 0.0f, racc = 0.0f, gacc = 0.0f, bacc = 0.0f, depth = 0.0f;

  for (int s = 0; s < N_SAMPLES; ++s) {
    const float z  = t_min + STEP * (float)s;
    const float px = ox + dx * z, py = oy + dy * z, pz = oz + dz * z;
    const bool valid = (px >= -1.5f) & (px <= 1.5f) &
                       (py >= -1.5f) & (py <= 1.5f) &
                       (pz >= -1.5f) & (pz <= 1.5f);
    const float xs = px * (2.0f / 3.0f);
    const float ys = py * (2.0f / 3.0f);
    const float zs = pz * (2.0f / 3.0f);

    // ---- Positional encoding: this lane covers k in [hi*18, hi*18+18) ----
    // pe index = c*12 + f (sin) or c*12 + 6 + f (cos), c in {x,y,z}, f in 0..5
    float pv[18];
#pragma unroll
    for (int i = 0; i < 18; ++i) {
      const int k = hi * 18 + i;        // constant after unroll? hi is runtime; k/12 etc fold per-lane-half
      const int c = k / 12;
      const int r = k % 12;
      const int f = (r >= 6) ? (r - 6) : r;
      const float x   = (c == 0) ? xs : ((c == 1) ? ys : zs);
      const float ang = x * (float)(1 << f);
      pv[i] = (r < 6) ? __sinf(ang) : __cosf(ang);
    }
#pragma unroll
    for (int i = 0; i < 9; ++i) {       // packed f16 pair stores (dword, aligned)
      h2 p; p.x = (_Float16)pv[2 * i]; p.y = (_Float16)pv[2 * i + 1];
      *(h2*)&peT[wave][row][hi * 18 + 2 * i] = p;
    }
    lds_fence();

    // ---- GEMM1: [16 rays x 64] @ [64 x 128] via 8 N-tiles x 2 K-steps ----
    const v16h a0 = load_a_frag(&peT[wave][0][0], 64, lane, 0);
    const v16h a1 = load_a_frag(&peT[wave][0][0], 64, lane, 1);
#pragma unroll
    for (int nt = 0; nt < 8; ++nt) {
      v8f c;
#pragma unroll
      for (int v = 0; v < 8; ++v) c[v] = biasv[nt];
      const v16h bf0 = *(const v16h*)&W1T[nt * 16 + row][hi * 16];
      const v16h bf1 = *(const v16h*)&W1T[nt * 16 + row][32 + hi * 16];
      c = __builtin_amdgcn_wmma_f32_16x16x32_f16(false, a0, false, bf0, (short)0, c, false, false);
      c = __builtin_amdgcn_wmma_f32_16x16x32_f16(false, a1, false, bf1, (short)0, c, false, false);
      // ReLU, convert, scatter to feat tile (C layout: M = v + 8*hi, N = nt*16 + row)
#pragma unroll
      for (int v = 0; v < 8; ++v) {
        const float f = fmaxf(c[v], 0.0f);
        featT[wave][v + 8 * hi][nt * 16 + row] = (_Float16)f;
      }
    }
    lds_fence();

    // ---- GEMM2 (head): [16 x 128] @ [128 x 16(pad 4)] via 4 K-steps ----
    v8f c2;
#pragma unroll
    for (int v = 0; v < 8; ++v) c2[v] = 0.0f;
#pragma unroll
    for (int ks = 0; ks < 4; ++ks) {
      const v16h af = load_a_frag(&featT[wave][0][0], 128, lane, ks);
      c2 = __builtin_amdgcn_wmma_f32_16x16x32_f16(false, af, false, b2[ks], (short)0, c2, false, false);
    }
    if (row < 4) {
#pragma unroll
      for (int v = 0; v < 8; ++v) outT[wave][v + 8 * hi][row] = c2[v];
    }
    lds_fence();

    // ---- Compositing scan (lanes 0-15 each own one ray) ----
    if (hi == 0) {
      const float sraw = outT[wave][row][0];
      const float r0 = outT[wave][row][1] + vt0;
      const float r1 = outT[wave][row][2] + vt1;
      const float r2 = outT[wave][row][3] + vt2;
      // softplus (stable)
      const float sigma   = fmaxf(sraw, 0.0f) + log1pf(__expf(-fabsf(sraw)));
      const float density = valid ? sigma : 0.0f;
      const float dist    = (s == N_SAMPLES - 1) ? 0.0f : STEP;
      const float alpha   = 1.0f - __expf(-density * dist * DIST_SCALE);
      const float w       = alpha * T;
      T = T * (1.0f - alpha + 1e-10f);
      const float sr = valid ? (1.0f / (1.0f + __expf(-r0))) : 0.0f;
      const float sg = valid ? (1.0f / (1.0f + __expf(-r1))) : 0.0f;
      const float sb = valid ? (1.0f / (1.0f + __expf(-r2))) : 0.0f;
      racc += w * sr; gacc += w * sg; bacc += w * sb;
      accw += w; depth += w * z;
    }
  }

  // ---- Finalize & write outputs ----
  if (hi == 0) {
    const float bg = 1.0f - accw;
    out[ray * 3 + 0] = fminf(fmaxf(racc + bg, 0.0f), 1.0f);
    out[ray * 3 + 1] = fminf(fmaxf(gacc + bg, 0.0f), 1.0f);
    out[ray * 3 + 2] = fminf(fmaxf(bacc + bg, 0.0f), 1.0f);
    out[N_RAYS * 3 + ray] = depth + bg * dz;   // rays_chunk[..., -1] == d.z
    out[N_RAYS * 4 + ray] = accw;
  }
}

extern "C" void kernel_launch(void* const* d_in, const int* in_sizes, int n_in,
                              void* d_out, int out_size, void* d_ws, size_t ws_size,
                              hipStream_t stream) {
  const float* rays  = (const float*)d_in[0];  // [8192,6]
  const float* W1    = (const float*)d_in[1];  // [36,128]
  const float* b1    = (const float*)d_in[2];  // [128]
  const float* w_d   = (const float*)d_in[3];  // [128]
  const float* W_rgb = (const float*)d_in[4];  // [131,3]
  float* out = (float*)d_out;                  // rgb | depth | acc
  (void)in_sizes; (void)n_in; (void)out_size; (void)d_ws; (void)ws_size;

  nerf_wmma_kernel<<<NBLOCKS, THREADS_PER_BLOCK, 0, stream>>>(
      rays, W1, b1, w_d, W_rgb, out);
}
